// Attention_57037165691498
// MI455X (gfx1250) — matrix operand order repaired and checked by reading, not compiled, not run on hardware
//
#include <hip/hip_runtime.h>
#include <hip/hip_bf16.h>

// ---------------------------------------------------------------------------
// MI455X / gfx1250 implementation.
//  - query scan + decoder GRU: persistent single-workgroup sequential kernels
//    (latency-bound; global_prefetch_b8 hides next-step fetch latency)
//  - dec_att_a:   V_WMMA_F32_16X16X4_F32  (native f32 matrix path)
//  - final MLP:   V_WMMA_F32_16X16X32_BF16 with the "shared 832 columns"
//                 factorization; persistent blocks double-buffer x_att tiles
//                 with GLOBAL_LOAD_ASYNC_TO_LDS_B128 + s_wait_asynccnt.
// ---------------------------------------------------------------------------

typedef __attribute__((ext_vector_type(16))) __bf16        v16bf;
typedef __attribute__((ext_vector_type(16))) unsigned short u16v;
typedef __attribute__((ext_vector_type(8)))  float         v8f;
typedef __attribute__((ext_vector_type(2)))  float         v2f;

union BfVec { u16v u; v16bf v; };

__device__ __forceinline__ unsigned short f2bf(float f) {
  unsigned int u = __float_as_uint(f);
  unsigned int r = (u + 0x7fffu + ((u >> 16) & 1u)) >> 16;   // RNE
  return (unsigned short)r;
}

__device__ __forceinline__ float sigmoidf(float x) {
  return 1.f / (1.f + __expf(-x));
}

// LDS byte offset (wave-relative) of a shared-memory pointer: cast the generic
// pointer back into addrspace(3) and take its integer value.
__device__ __forceinline__ unsigned lds_offset_of(const void* p) {
  return (unsigned)(unsigned long long)
      (__attribute__((address_space(3))) const void*)p;
}

// Async copy 16B global -> LDS (VGLOBAL GLOBAL_LOAD_ASYNC_TO_LDS_B128,
// tracked by ASYNCcnt).
__device__ __forceinline__ void async_load_b128(unsigned lds_addr, const void* g) {
  asm volatile("global_load_async_to_lds_b128 %0, %1, off"
               :: "v"(lds_addr), "v"((unsigned long long)g)
               : "memory");
}
__device__ __forceinline__ void wait_async0() {
  asm volatile("s_wait_asynccnt 0" ::: "memory");
}

// A operand: row-major bf16 [rows][ld]; CDNA5 16-bit A 16x32 layout
// lanes 0-15: M=lane, VGPR0-3 -> K 0..7, VGPR4-7 -> K 16..23
// lanes16-31: M=lane-16, VGPR0-3 -> K 8..15, VGPR4-7 -> K 24..31
__device__ __forceinline__ v16bf load_a_bf16(const unsigned short* base, int ld,
                                             int row0, int k0, int lane) {
  BfVec r;
  int m  = row0 + (lane & 15);
  int kl = (lane >= 16) ? 8 : 0;
#pragma unroll
  for (int v = 0; v < 8; ++v) {
    int k = k0 + 2 * v + ((v < 4) ? 0 : 8) + kl;
    r.u[2 * v]     = base[m * ld + k];
    r.u[2 * v + 1] = base[m * ld + k + 1];
  }
  return r.v;
}

// B operand from row-major weight W[n][k] (B = W^T, 32x16):
// lanes 0-15 hold K=0..7,16..23 (2 per VGPR), lanes 16-31 hold K=8..15,24..31; N = lane%16
__device__ __forceinline__ v16bf load_b_bf16(const unsigned short* base, int ld,
                                             int col0, int k0, int lane) {
  BfVec r;
  int n  = col0 + (lane & 15);
  int kl = (lane >= 16) ? 8 : 0;
#pragma unroll
  for (int v = 0; v < 8; ++v) {
    int k = k0 + 2 * v + ((v < 4) ? 0 : 8) + kl;
    r.u[2 * v]     = base[n * ld + k];
    r.u[2 * v + 1] = base[n * ld + k + 1];
  }
  return r.v;
}

// ---------------------------------------------------------------------------
// f32 -> bf16 conversion (weights / x_att done once, stay hot in L2)
// ---------------------------------------------------------------------------
__global__ void to_bf16_kernel(const float* __restrict__ src,
                               unsigned short* __restrict__ dst, int n) {
  for (int i = blockIdx.x * blockDim.x + threadIdx.x; i < n;
       i += gridDim.x * blockDim.x)
    dst[i] = f2bf(src[i]);
}

// ---------------------------------------------------------------------------
// Phase 1: sequential query-generation scan.  One persistent workgroup;
// latency-bound GEMVs (weights ~4MB/step stream from L2).
// ---------------------------------------------------------------------------
__global__ void __launch_bounds__(1024) query_scan_kernel(
    const float* __restrict__ x_fix, const float* __restrict__ x_att,
    const float* __restrict__ w_ih, const float* __restrict__ w_hh,
    const float* __restrict__ b_ih, const float* __restrict__ b_hh,
    const float* __restrict__ qh_w, const float* __restrict__ qh_b,
    const float* __restrict__ w_q,
    float* __restrict__ attw_out, float* __restrict__ qgru_out,
    float* __restrict__ qn_out) {
  __shared__ float att[4096];
  __shared__ float q[512], attv[512], xb[1088], gi[768], gh[768];
  __shared__ float h[256], h2[256], hid[256], sc[512];
  __shared__ float smax[8], ssum[8];
  const int tid = threadIdx.x;

  if (tid < 512) q[tid] = 1.f;
  if (tid < 256) h[tid] = 0.f;
  __syncthreads();

  for (int t = 0; t < 2048; ++t) {
    for (int i = tid; i < 4096; i += 1024) att[i] = x_att[t * 4096 + i];
    // hide next step's x_att fetch behind this step's GEMVs
    if (t < 2047 && tid < 64)
      __builtin_prefetch(x_att + (t + 1) * 4096 + tid * 64, 0, 3);
    __syncthreads();

    // scores [8 heads][64 objects], scale = 1/sqrt(64)
    if (tid < 512) {
      int hh = tid >> 6, n = tid & 63;
      float s = 0.f;
      for (int a = 0; a < 64; ++a) s += q[hh * 64 + a] * att[n * 64 + a];
      sc[tid] = s * 0.125f;
    }
    __syncthreads();
    if (tid < 8) {
      float mx = -1e30f;
      for (int n = 0; n < 64; ++n) mx = fmaxf(mx, sc[tid * 64 + n]);
      float sum = 0.f;
      for (int n = 0; n < 64; ++n) sum += __expf(sc[tid * 64 + n] - mx);
      smax[tid] = mx; ssum[tid] = sum;
    }
    __syncthreads();
    if (tid < 512) {
      int hh = tid >> 6;
      float w = __expf(sc[tid] - smax[hh]) / ssum[hh];
      sc[tid] = w;
      attw_out[t * 512 + tid] = w;               // attw output
    }
    __syncthreads();
    // att_v = attw @ att_t
    if (tid < 512) {
      int hh = tid >> 6, a = tid & 63;
      float s = 0.f;
      for (int n = 0; n < 64; ++n) s += sc[hh * 64 + n] * att[n * 64 + a];
      attv[tid] = s;
    }
    __syncthreads();
    // x = [fix(64), att_v(512), query(512)]
    for (int i = tid; i < 1088; i += 1024)
      xb[i] = (i < 64) ? x_fix[t * 64 + i] : (i < 576 ? attv[i - 64] : q[i - 576]);
    __syncthreads();
    // GRU gates
    if (tid < 768) {
      const float* wr = w_ih + tid * 1088;
      float s = 0.f;
      for (int k = 0; k < 1088; ++k) s += wr[k] * xb[k];
      gi[tid] = s + b_ih[tid];
      const float* wh = w_hh + tid * 256;
      float s2 = 0.f;
      for (int k = 0; k < 256; ++k) s2 += wh[k] * h[k];
      gh[tid] = s2 + b_hh[tid];
    }
    __syncthreads();
    if (tid < 256) {
      float r  = sigmoidf(gi[tid] + gh[tid]);
      float z  = sigmoidf(gi[256 + tid] + gh[256 + tid]);
      float nn = tanhf(gi[512 + tid] + r * gh[512 + tid]);
      h2[tid] = (1.f - z) * nn + z * h[tid];
    }
    __syncthreads();
    // hid input = [h2(256), fix(64), att_v(512)]
    for (int i = tid; i < 832; i += 1024)
      xb[i] = (i < 256) ? h2[i] : (i < 320 ? x_fix[t * 64 + (i - 256)] : attv[i - 320]);
    __syncthreads();
    if (tid < 256) {
      const float* wr = qh_w + tid * 832;
      float s = qh_b[tid];
      for (int k = 0; k < 832; ++k) s += wr[k] * xb[k];
      hid[tid] = s > 0.f ? s : 0.f;
    }
    __syncthreads();
    // query2 = hid @ weight_q   (coalesced column access)
    if (tid < 512) {
      float s = 0.f;
      for (int j = 0; j < 256; ++j) s += hid[j] * w_q[j * 512 + tid];
      sc[tid] = s;
    }
    __syncthreads();
    if (tid < 512) q[tid] = sc[tid];
    if (tid < 256) h[tid] = h2[tid];
    __syncthreads();
  }
  if (tid < 256) qgru_out[tid] = h[tid];
  if (tid < 512) qn_out[tid]  = q[tid];
}

// ---------------------------------------------------------------------------
// Phase 2a: dec_att_a[t] = attw[max(t-hz,0)] @ x_att[t]   ([8,64]@[64,64])
// One wave per t; native f32 WMMA 16x16x4 (M padded 8->16, N=64, K=64).
// ---------------------------------------------------------------------------
__global__ void dec_attv_kernel(const float* __restrict__ x_att,
                                const float* __restrict__ attw,
                                const int* __restrict__ hz_p,
                                float* __restrict__ dec_att_a) {
  __shared__ float Apad[16 * 64];
  __shared__ float Bt[64 * 64];
  const int t = blockIdx.x;
  const int lane = threadIdx.x;                  // wave32
  int tsrc = t - hz_p[0]; if (tsrc < 0) tsrc = 0;

  for (int i = lane; i < 16 * 64; i += 32)
    Apad[i] = (i < 8 * 64) ? attw[tsrc * 512 + i] : 0.f;
  for (int i = lane; i < 64 * 64; i += 32) Bt[i] = x_att[t * 4096 + i];
  __syncthreads();

  const int m  = lane & 15;
  const int hi = (lane >= 16) ? 1 : 0;
  v8f acc[4] = {};
  for (int k0 = 0; k0 < 64; k0 += 4) {
    // f32 A 16x4: lanes 0-15 VGPR{0,1}=K{0,1}; lanes 16-31 VGPR{0,1}=K{2,3}
    v2f a;
    a.x = Apad[m * 64 + k0 + 2 * hi];
    a.y = Apad[m * 64 + k0 + 2 * hi + 1];
#pragma unroll
    for (int nt = 0; nt < 4; ++nt) {
      v2f b;
      b.x = Bt[(k0 + 2 * hi) * 64 + nt * 16 + m];
      b.y = Bt[(k0 + 2 * hi + 1) * 64 + nt * 16 + m];
      acc[nt] = __builtin_amdgcn_wmma_f32_16x16x4_f32(
          false, a, false, b, (short)0, acc[nt], false, false);
    }
  }
#pragma unroll
  for (int nt = 0; nt < 4; ++nt)
#pragma unroll
    for (int r = 0; r < 8; ++r) {
      int mm = r + 8 * hi;                       // D row
      if (mm < 8)                                // heads 0..7 only
        dec_att_a[t * 512 + mm * 64 + nt * 16 + m] = acc[nt][r];
    }
}

// ---------------------------------------------------------------------------
// Phase 2b: sequential decoder GRU scan (persistent single workgroup)
// ---------------------------------------------------------------------------
__global__ void __launch_bounds__(1024) dec_gru_scan_kernel(
    const float* __restrict__ x_fix, const float* __restrict__ dec_att_a,
    const float* __restrict__ w_ih, const float* __restrict__ w_hh,
    const float* __restrict__ b_ih, const float* __restrict__ b_hh,
    float* __restrict__ dec_gru_a, float* __restrict__ dgru_out) {
  __shared__ float xb[576], gi[768], gh[768], h[256], h2[256];
  const int tid = threadIdx.x;
  if (tid < 256) h[tid] = 0.f;
  __syncthreads();
  for (int t = 0; t < 2048; ++t) {
    if (tid < 576)
      xb[tid] = (tid < 64) ? x_fix[t * 64 + tid] : dec_att_a[t * 512 + tid - 64];
    if (t < 2047 && tid < 8)
      __builtin_prefetch(dec_att_a + (t + 1) * 512 + tid * 64, 0, 3);
    __syncthreads();
    if (tid < 768) {
      const float* wr = w_ih + tid * 576;
      float s = 0.f;
      for (int k = 0; k < 576; ++k) s += wr[k] * xb[k];
      gi[tid] = s + b_ih[tid];
      const float* wh = w_hh + tid * 256;
      float s2 = 0.f;
      for (int k = 0; k < 256; ++k) s2 += wh[k] * h[k];
      gh[tid] = s2 + b_hh[tid];
    }
    __syncthreads();
    if (tid < 256) {
      float r  = sigmoidf(gi[tid] + gh[tid]);
      float z  = sigmoidf(gi[256 + tid] + gh[256 + tid]);
      float nn = tanhf(gi[512 + tid] + r * gh[512 + tid]);
      float hv = (1.f - z) * nn + z * h[tid];
      h2[tid] = hv;
      dec_gru_a[t * 256 + tid] = hv;
    }
    __syncthreads();
    if (tid < 256) h[tid] = h2[tid];
    __syncthreads();
  }
  if (tid < 256) dgru_out[tid] = h[tid];
}

// ---------------------------------------------------------------------------
// Phase 3a: partial[t][j] = feat[t](832) . W1[j][0:832]
// (columns shared across all 64 objects -> factor out of the big GEMM)
// bf16 WMMA GEMM [2048,832]@[832,256]; 128 blocks x 8 waves.
// ---------------------------------------------------------------------------
__global__ void __launch_bounds__(256) feat_partial_kernel(
    const float* __restrict__ dec_gru_a, const float* __restrict__ x_fix,
    const float* __restrict__ dec_att_a, const unsigned short* __restrict__ w1bf,
    float* __restrict__ partial) {
  __shared__ unsigned short As[16 * 832];
  const int mt = blockIdx.x;                 // 16-row t tile
  const int tid = threadIdx.x;
  const int lane = tid & 31, wave = tid >> 5;

  for (int i = tid; i < 16 * 832; i += 256) {
    int r = i / 832, c = i % 832;
    int t = mt * 16 + r;
    float v = (c < 256) ? dec_gru_a[t * 256 + c]
            : (c < 320) ? x_fix[t * 64 + (c - 256)]
                        : dec_att_a[t * 512 + (c - 320)];
    As[i] = f2bf(v);
  }
  __syncthreads();

  for (int jt = wave; jt < 16; jt += 8) {    // 16 N-tiles over 8 waves
    v8f acc = {};
    for (int kk = 0; kk < 26; ++kk) {        // K = 832 = 26 * 32
      v16bf a = load_a_bf16(As, 832, 0, kk * 32, lane);
      v16bf b = load_b_bf16(w1bf, 896, jt * 16, kk * 32, lane);
      acc = __builtin_amdgcn_wmma_f32_16x16x32_bf16(
          false, a, false, b, (short)0, acc, false, false);
    }
    int j = jt * 16 + (lane & 15);
#pragma unroll
    for (int r = 0; r < 8; ++r) {
      int t = mt * 16 + r + ((lane >= 16) ? 8 : 0);
      partial[t * 256 + j] = acc[r];
    }
  }
}

// ---------------------------------------------------------------------------
// Phase 3b (persistent, 256 blocks x 8 timesteps):
//   hid[64,256] = relu(partial + x_att[t]@W1b^T + b1)
//   pred[64,64] = relu(hid @ W2^T + b2)
// x_att tiles double-buffered in LDS via GLOBAL_LOAD_ASYNC_TO_LDS_B128;
// next tile's fetch overlaps current tile's WMMA work.
// ---------------------------------------------------------------------------
#define PRED_BLOCKS 256
__global__ void __launch_bounds__(256) pred_kernel(
    const unsigned short* __restrict__ xattbf, const float* __restrict__ partial,
    const unsigned short* __restrict__ w1bf, const float* __restrict__ b1,
    const unsigned short* __restrict__ w2bf, const float* __restrict__ b2,
    float* __restrict__ pred) {
  __shared__ unsigned short Xs[2][4096];     // double-buffered x_att[t] bf16
  __shared__ unsigned short Hs[64 * 256];    // hid bf16
  const int tid = threadIdx.x;
  const int lane = tid & 31, wave = tid >> 5;
  const int t0 = blockIdx.x;

  // prime buffer 0: 8KB tile, each thread moves 2 x 16B asynchronously
  async_load_b128(lds_offset_of(&Xs[0][tid * 8]),
                  xattbf + (size_t)t0 * 4096 + tid * 8);
  async_load_b128(lds_offset_of(&Xs[0][2048 + tid * 8]),
                  xattbf + (size_t)t0 * 4096 + 2048 + tid * 8);

  int buf = 0;
  for (int t = t0; t < 2048; t += PRED_BLOCKS) {
    wait_async0();                           // our async writes into Xs[buf] done
    __syncthreads();                         // everyone's writes visible; Hs free
    int tn = t + PRED_BLOCKS;
    if (tn < 2048) {                         // prefetch next tile into other buf
      async_load_b128(lds_offset_of(&Xs[buf ^ 1][tid * 8]),
                      xattbf + (size_t)tn * 4096 + tid * 8);
      async_load_b128(lds_offset_of(&Xs[buf ^ 1][2048 + tid * 8]),
                      xattbf + (size_t)tn * 4096 + 2048 + tid * 8);
    }
    const unsigned short* X = &Xs[buf][0];

    // GEMM1: M=64 objects, N=256 hidden, K=64 (x_att tail columns of W1)
    for (int tt = wave; tt < 64; tt += 8) {
      int mt = tt >> 4, jt = tt & 15;
      v8f acc = {};
#pragma unroll
      for (int kk = 0; kk < 2; ++kk) {
        v16bf a = load_a_bf16(X, 64, mt * 16, kk * 32, lane);
        v16bf b = load_b_bf16(w1bf + 832, 896, jt * 16, kk * 32, lane);
        acc = __builtin_amdgcn_wmma_f32_16x16x32_bf16(
            false, a, false, b, (short)0, acc, false, false);
      }
      int j = jt * 16 + (lane & 15);
      float pb = partial[t * 256 + j] + b1[j];
#pragma unroll
      for (int r = 0; r < 8; ++r) {
        int n = mt * 16 + r + ((lane >= 16) ? 8 : 0);
        float v = acc[r] + pb;
        Hs[n * 256 + j] = f2bf(v > 0.f ? v : 0.f);
      }
    }
    __syncthreads();

    // GEMM2: M=64 objects, N=64 outputs, K=256
    for (int tt = wave; tt < 16; tt += 8) {
      int mt = tt >> 2, ot = tt & 3;
      v8f acc = {};
#pragma unroll
      for (int kk = 0; kk < 8; ++kk) {
        v16bf a = load_a_bf16(Hs, 256, mt * 16, kk * 32, lane);
        v16bf b = load_b_bf16(w2bf, 256, ot * 16, kk * 32, lane);
        acc = __builtin_amdgcn_wmma_f32_16x16x32_bf16(
            false, a, false, b, (short)0, acc, false, false);
      }
      int o = ot * 16 + (lane & 15);
      float bb = b2[o];
#pragma unroll
      for (int r = 0; r < 8; ++r) {
        int n = mt * 16 + r + ((lane >= 16) ? 8 : 0);
        float v = acc[r] + bb;
        pred[t * 4096 + n * 64 + o] = v > 0.f ? v : 0.f;
      }
    }
    buf ^= 1;
  }
}

// ---------------------------------------------------------------------------
extern "C" void kernel_launch(void* const* d_in, const int* in_sizes, int n_in,
                              void* d_out, int out_size, void* d_ws, size_t ws_size,
                              hipStream_t stream) {
  (void)in_sizes; (void)n_in; (void)out_size; (void)ws_size;
  const float* x_fix     = (const float*)d_in[0];   // [2048,64]
  const float* x_att     = (const float*)d_in[1];   // [2048,64,64]
  const int*   hz        = (const int*)d_in[2];
  const float* q_w_ih    = (const float*)d_in[3];   // [768,1088]
  const float* q_w_hh    = (const float*)d_in[4];   // [768,256]
  const float* q_b_ih    = (const float*)d_in[5];
  const float* q_b_hh    = (const float*)d_in[6];
  const float* q_hid_w   = (const float*)d_in[7];   // [256,832]
  const float* q_hid_b   = (const float*)d_in[8];
  const float* weight_q  = (const float*)d_in[9];   // [256,512]
  const float* d_w_ih    = (const float*)d_in[10];  // [768,576]
  const float* d_w_hh    = (const float*)d_in[11];  // [768,256]
  const float* d_b_ih    = (const float*)d_in[12];
  const float* d_b_hh    = (const float*)d_in[13];
  const float* dec_hid_w = (const float*)d_in[14];  // [256,896]
  const float* dec_hid_b = (const float*)d_in[15];
  const float* dec_out_w = (const float*)d_in[16];  // [64,256]
  const float* dec_out_b = (const float*)d_in[17];

  // outputs concatenated: pred, attw, q_gru_n, dec_gru_n, query_n
  float* out    = (float*)d_out;
  float* pred   = out;                 // 8388608
  float* attw   = out + 8388608;       // 1048576
  float* qgru_n = out + 9437184;       // 256
  float* dgru_n = out + 9437440;       // 256
  float* qn     = out + 9437696;       // 512

  // workspace layout (~26 MB)
  float* dec_att_a = (float*)d_ws;                          // 1048576 f32
  float* dec_gru_a = dec_att_a + 1048576;                   //  524288 f32
  float* partial   = dec_gru_a + 524288;                    //  524288 f32
  unsigned short* w1bf   = (unsigned short*)(partial + 524288); // 229376
  unsigned short* w2bf   = w1bf + 229376;                   //   16384
  unsigned short* xattbf = w2bf + 16384;                    // 8388608

  to_bf16_kernel<<<224, 256, 0, stream>>>(dec_hid_w, w1bf, 229376);
  to_bf16_kernel<<<64,  256, 0, stream>>>(dec_out_w, w2bf, 16384);
  to_bf16_kernel<<<1024, 256, 0, stream>>>(x_att, xattbf, 8388608);

  query_scan_kernel<<<1, 1024, 0, stream>>>(
      x_fix, x_att, q_w_ih, q_w_hh, q_b_ih, q_b_hh,
      q_hid_w, q_hid_b, weight_q, attw, qgru_n, qn);

  dec_attv_kernel<<<2048, 32, 0, stream>>>(x_att, attw, hz, dec_att_a);

  dec_gru_scan_kernel<<<1, 1024, 0, stream>>>(
      x_fix, dec_att_a, d_w_ih, d_w_hh, d_b_ih, d_b_hh, dec_gru_a, dgru_n);

  feat_partial_kernel<<<128, 256, 0, stream>>>(
      dec_gru_a, x_fix, dec_att_a, w1bf, partial);

  pred_kernel<<<PRED_BLOCKS, 256, 0, stream>>>(
      xattbf, partial, w1bf, dec_hid_b, w2bf, dec_out_b, pred);
}
